// HSMMLayer_18279380812095
// MI455X (gfx1250) — matrix operand order repaired and checked by reading, not compiled, not run on hardware
//
#include <hip/hip_runtime.h>
#include <hip/hip_bf16.h>
#include <math.h>

typedef __attribute__((ext_vector_type(16))) _Float16 v16h;
typedef __attribute__((ext_vector_type(8)))  float    v8f;

#define B_  8
#define T_  800
#define S_  64
#define F_  80
#define D_  50
#define K_  160              // [x^2 | x]
#define RING 52              // csum ring: 51-row window + 1 prefetch slot
#define NEG (-1e30f)
#define EPSF (1e-8f)

// workspace layout in float words
#define TRANS_OFF 0          // 64*64
#define DUR_OFF   4096       // 64*50
#define CONST_OFF 7296       // 64
#define W_OFF     7360       // 64*160 halves (layout [s][k]) = 5120 float words
#define OBSLP_OFF 12480      // 6400*64
#define CSUM_OFF  422080     // 8*801*64
#define BP_OFF    832192     // 8*800*64 int
#define ARGD_OFF  1241792    // 8*800*64 int
#define FIN_OFF   1651392    // 8*3 words: s0, d0, score

__device__ __forceinline__ float softplus_f(float x) {
  return (x > 20.f) ? x : log1pf(expf(x));
}

// low 32 bits of a flat shared-aperture address == LDS byte offset
__device__ __forceinline__ unsigned lds_off_of(const void* p) {
  return (unsigned)(uintptr_t)p;
}

__device__ __forceinline__ void async_load_lds_b32(unsigned lds, const void* gaddr) {
  asm volatile("global_load_async_to_lds_b32 %0, %1, off"
               :: "v"(lds), "v"(gaddr) : "memory");
}
__device__ __forceinline__ void async_load_lds_b128(unsigned lds, const void* gaddr) {
  asm volatile("global_load_async_to_lds_b128 %0, %1, off"
               :: "v"(lds), "v"(gaddr) : "memory");
}
__device__ __forceinline__ void wait_async0() {
  asm volatile("s_wait_asynccnt 0x0" ::: "memory");
}

// ---------------------------------------------------------------------------
// Kernel 1: tiny precompute — log-transition matrix (diag = NEG), gamma
// duration log-probs, per-state fp32 constant, fp16 GEMM weights W[s][k].
// ---------------------------------------------------------------------------
__global__ void __launch_bounds__(256) precompute_kernel(
    const float* __restrict__ logits, const float* __restrict__ means,
    const float* __restrict__ lvars, const float* __restrict__ shp,
    const float* __restrict__ rte,
    float* __restrict__ trans, float* __restrict__ dur,
    float* __restrict__ cst, _Float16* __restrict__ W) {
  const int tid = threadIdx.x;
  if (tid < 64) {                       // transition row tid
    const int p = tid;
    float mx = -3.4e38f;
    for (int q = 0; q < S_; ++q) if (q != p) mx = fmaxf(mx, logits[p * S_ + q]);
    float sum = 0.f;
    for (int q = 0; q < S_; ++q) if (q != p) sum += expf(logits[p * S_ + q] - mx);
    for (int q = 0; q < S_; ++q)
      trans[p * S_ + q] = (q == p) ? NEG
                                   : logf(expf(logits[p * S_ + q] - mx) / sum + EPSF);
  } else if (tid < 128) {               // duration log-probs for state tid-64
    const int si = tid - 64;
    const float a = softplus_f(shp[si]);
    const float r = softplus_f(rte[si]);
    const float lga = lgammaf(a);
    const float alr = a * logf(r + EPSF);
    for (int d = 1; d <= D_; ++d) {
      float lp = (a - 1.f) * logf((float)d + EPSF) - r * (float)d - lga + alr;
      dur[si * D_ + (d - 1)] = logf(expf(lp) + EPSF);
    }
  } else if (tid < 192) {               // fp32 constant term for state tid-128
    const int si = tid - 128;
    float c = 0.f;
    for (int f = 0; f < F_; ++f) {
      float lv = lvars[si * F_ + f];
      float m  = means[si * F_ + f];
      c += m * m * expf(-lv) + lv;
    }
    cst[si] = c + (float)F_ * 1.8378770664093453f;   // F*log(2*pi)
  }
  // W[s][k]: k<80 -> -0.5*inv_var[s,k] ; k>=80 -> mean*inv_var at f=k-80
  for (int idx = tid; idx < S_ * K_; idx += 256) {
    const int si = idx / K_, k = idx % K_;
    float w;
    if (k < F_) {
      w = -0.5f * expf(-lvars[si * F_ + k]);
    } else {
      const int f = k - F_;
      w = means[si * F_ + f] * expf(-lvars[si * F_ + f]);
    }
    W[idx] = (_Float16)w;
  }
}

// ---------------------------------------------------------------------------
// Kernel 2: obs_lp GEMM via WMMA. W staged block-wide into LDS with async
// b128 copies; B fragments become contiguous 32B LDS reads. A row =
// [x^2 | x] built on the fly in fp16; fp32 accumulate; const added at store.
// ---------------------------------------------------------------------------
__global__ void __launch_bounds__(256) obslp_wmma_kernel(
    const float* __restrict__ obs, const _Float16* __restrict__ W,
    const float* __restrict__ cst, float* __restrict__ obs_lp) {
  __shared__ __align__(32) _Float16 l_W[S_ * K_];   // 20KB, [s][k]
  // async stage W: 10240 halves = 1280 x b128
  for (int c = threadIdx.x; c < (S_ * K_) / 8; c += 256)
    async_load_lds_b128(lds_off_of(&l_W[c * 8]), W + c * 8);
  wait_async0();
  __syncthreads();

  const int lane = threadIdx.x & 31;
  const int wave = threadIdx.x >> 5;
  const int mt   = blockIdx.x * 8 + wave;          // 0..399 M-tiles
  const int rowA = mt * 16 + (lane & 15);
  const int ka_off = (lane >> 4) << 4;             // A frag: +16 K for hi lanes
  const int kb_off = (lane >> 4) << 4;             // B frag: +16 K for hi lanes
  const int colB = lane & 15;
  const float* orow = obs + (size_t)rowA * F_;

  v8f acc0 = {}, acc1 = {}, acc2 = {}, acc3 = {};
  for (int kb = 0; kb < K_; kb += 32) {
    v16h a;
#pragma unroll
    for (int i = 0; i < 16; ++i) {
      const int k = kb + ka_off + i;
      float x;
      if (k < F_) { float o = orow[k]; x = o * o; }
      else        { x = orow[k - F_]; }
      a[i] = (_Float16)x;
    }
    // contiguous 32B LDS reads per fragment (ds_load_b128 pairs)
    const v16h b0 = *(const v16h*)&l_W[(colB +  0) * K_ + kb + kb_off];
    const v16h b1 = *(const v16h*)&l_W[(colB + 16) * K_ + kb + kb_off];
    const v16h b2 = *(const v16h*)&l_W[(colB + 32) * K_ + kb + kb_off];
    const v16h b3 = *(const v16h*)&l_W[(colB + 48) * K_ + kb + kb_off];
    acc0 = __builtin_amdgcn_wmma_f32_16x16x32_f16(false, a, false, b0, (short)0, acc0, false, false);
    acc1 = __builtin_amdgcn_wmma_f32_16x16x32_f16(false, a, false, b1, (short)0, acc1, false, false);
    acc2 = __builtin_amdgcn_wmma_f32_16x16x32_f16(false, a, false, b2, (short)0, acc2, false, false);
    acc3 = __builtin_amdgcn_wmma_f32_16x16x32_f16(false, a, false, b3, (short)0, acc3, false, false);
  }
  const int rowOff = (lane >> 4) << 3;             // C/D layout: hi lanes = M+8
#pragma unroll
  for (int n = 0; n < 4; ++n) {
    const v8f acc = (n == 0) ? acc0 : (n == 1) ? acc1 : (n == 2) ? acc2 : acc3;
    const int col = n * 16 + colB;
    const float c = cst[col];
#pragma unroll
    for (int v = 0; v < 8; ++v) {
      const int row = mt * 16 + v + rowOff;
      obs_lp[(size_t)row * S_ + col] = acc[v] - 0.5f * c;
    }
  }
}

// ---------------------------------------------------------------------------
// Kernel 3: per-(b,s) prefix sums along T (512 independent sequences).
// ---------------------------------------------------------------------------
__global__ void __launch_bounds__(256) csum_kernel(
    const float* __restrict__ obs_lp, float* __restrict__ csum) {
  const int idx = blockIdx.x * 256 + threadIdx.x;
  if (idx >= B_ * S_) return;
  const int b = idx >> 6, s = idx & 63;
  float acc = 0.f;
  csum[(size_t)(b * (T_ + 1)) * S_ + s] = 0.f;
  for (int t = 0; t < T_; ++t) {
    acc += obs_lp[((size_t)b * T_ + t) * S_ + s];
    csum[((size_t)b * (T_ + 1) + t + 1) * S_ + s] = acc;
  }
}

// ---------------------------------------------------------------------------
// Kernel 4: semi-Markov Viterbi scan. One block per batch chain; all state
// LDS-resident; cumsum rows prefetched 2 steps ahead with async-to-LDS so the
// sequential chain never waits on L2.
// ---------------------------------------------------------------------------
__global__ void __launch_bounds__(256) viterbi_kernel(
    const float* __restrict__ csum, const float* __restrict__ trans,
    const float* __restrict__ dur, int* __restrict__ bp,
    int* __restrict__ argd_g, float* __restrict__ fin) {
  __shared__ float l_trans[S_ * S_];
  __shared__ float l_dur[S_ * D_];
  __shared__ float l_csum[RING * S_];       // slot r%RING holds csum row r
  __shared__ float l_hist[D_ * S_];         // ring: slot tau%D holds M[tau]
  __shared__ float l_A[S_];
  __shared__ int   l_argd[S_];
  const int b = blockIdx.x;
  const int tid = threadIdx.x;

  for (int i = tid; i < S_ * S_; i += 256) l_trans[i] = trans[i];
  for (int i = tid; i < S_ * D_; i += 256) l_dur[i] = dur[i];
  for (int i = tid; i < D_ * S_; i += 256) l_hist[i] = (i >= (D_ - 1) * S_) ? 0.f : NEG;
  if (tid < 128) {                          // preload csum rows 0 and 1
    const int r = tid >> 6, s0 = tid & 63;
    l_csum[r * S_ + s0] = csum[((size_t)b * (T_ + 1) + r) * S_ + s0];
  }
  __syncthreads();

  const int s = tid >> 2;
  const int j = tid & 3;
  for (int e = 0; e < T_; ++e) {
    // row e+1 was async-prefetched during iteration e-1 (or preloaded)
    if (tid < S_) wait_async0();
    __syncthreads();
    // prefetch row e+2 into the slot of row e-50 (outside this step's window)
    if (tid < S_ && (e + 2) <= T_)
      async_load_lds_b32(lds_off_of(&l_csum[((e + 2) % RING) * S_ + tid]),
                         &csum[((size_t)b * (T_ + 1) + e + 2) * S_ + tid]);

    // phase 1: best duration per state
    const float ce1 = l_csum[((e + 1) % RING) * S_ + s];
    float best = NEG; int bd = 0;
    for (int d = j + 1; d <= D_; d += 4) {
      const int tau = e - d;
      const int slot = (tau >= 0) ? (tau % D_) : (tau + D_);
      int st = tau + 1; if (st < 0) st = 0;
      const float v = l_hist[slot * S_ + s] +
                      (ce1 - l_csum[(st % RING) * S_ + s]) +
                      l_dur[s * D_ + (d - 1)];
      if (v > best) { best = v; bd = d - 1; }
    }
#pragma unroll
    for (int off = 1; off < 4; off <<= 1) {
      const float ob = __shfl_xor(best, off, 32);
      const int   od = __shfl_xor(bd, off, 32);
      if (ob > best || (ob == best && od < bd)) { best = ob; bd = od; }
    }
    if (j == 0) { l_A[s] = best; l_argd[s] = bd; }
    __syncthreads();

    // phase 2: best previous state (max-plus with transition matrix)
    float bm = NEG; int bprev = 0;
    const int p0 = j * 16;
    for (int p = p0; p < p0 + 16; ++p) {
      const float v = l_A[p] + l_trans[p * S_ + s];
      if (v > bm) { bm = v; bprev = p; }
    }
#pragma unroll
    for (int off = 1; off < 4; off <<= 1) {
      const float ob = __shfl_xor(bm, off, 32);
      const int   op = __shfl_xor(bprev, off, 32);
      if (ob > bm || (ob == bm && op < bprev)) { bm = ob; bprev = op; }
    }
    if (j == 0) {
      l_hist[(e % D_) * S_ + s] = bm;                       // M[e]
      bp[((size_t)b * T_ + e) * S_ + s]     = bprev;
      argd_g[((size_t)b * T_ + e) * S_ + s] = l_argd[s];
    }
    __syncthreads();
  }

  if (tid == 0) {       // flat argmax over delta[T-1] == (argmax_s A, argd[s])
    float sc = l_A[0]; int s0 = 0;
    for (int i = 1; i < S_; ++i) if (l_A[i] > sc) { sc = l_A[i]; s0 = i; }
    ((int*)fin)[b * 3 + 0] = s0;
    ((int*)fin)[b * 3 + 1] = l_argd[s0] + 1;
    fin[b * 3 + 2] = sc;
  }
}

// ---------------------------------------------------------------------------
// Kernel 5: backtracking — one thread per batch element.
// ---------------------------------------------------------------------------
__global__ void backtrack_kernel(const int* __restrict__ bp,
                                 const int* __restrict__ argd_g,
                                 const float* __restrict__ fin,
                                 float* __restrict__ out) {
  const int b = threadIdx.x;
  if (b >= B_) return;
  const int* fi = (const int*)fin;
  int s = fi[b * 3 + 0];
  int d = fi[b * 3 + 1];
  out[(size_t)B_ * T_ + b] = fin[b * 3 + 2];   // score
  int t = T_ - 1;
  while (t >= 0) {
    int lo = t - d + 1; if (lo < 0) lo = 0;
    for (int i = lo; i <= t; ++i) out[(size_t)b * T_ + i] = (float)s;
    const int tau = t - d;
    if (tau < 0) break;
    const int ps = bp[((size_t)b * T_ + tau) * S_ + s];
    d = argd_g[((size_t)b * T_ + tau) * S_ + ps] + 1;
    s = ps;
    t = tau;
  }
}

// ---------------------------------------------------------------------------
extern "C" void kernel_launch(void* const* d_in, const int* in_sizes, int n_in,
                              void* d_out, int out_size, void* d_ws, size_t ws_size,
                              hipStream_t stream) {
  (void)in_sizes; (void)n_in; (void)out_size; (void)ws_size;
  const float* obs    = (const float*)d_in[0];
  const float* logits = (const float*)d_in[1];
  const float* means  = (const float*)d_in[2];
  const float* lvars  = (const float*)d_in[3];
  const float* shp    = (const float*)d_in[4];
  const float* rte    = (const float*)d_in[5];

  float* ws       = (float*)d_ws;
  float* trans    = ws + TRANS_OFF;
  float* dur      = ws + DUR_OFF;
  float* cst      = ws + CONST_OFF;
  _Float16* Wf16  = (_Float16*)(ws + W_OFF);
  float* obslp    = ws + OBSLP_OFF;
  float* csum     = ws + CSUM_OFF;
  int*   bp       = (int*)(ws + BP_OFF);
  int*   argd     = (int*)(ws + ARGD_OFF);
  float* fin      = ws + FIN_OFF;

  precompute_kernel<<<1, 256, 0, stream>>>(logits, means, lvars, shp, rte,
                                           trans, dur, cst, Wf16);
  obslp_wmma_kernel<<<50, 256, 0, stream>>>(obs, Wf16, cst, obslp);
  csum_kernel<<<2, 256, 0, stream>>>(obslp, csum);
  viterbi_kernel<<<B_, 256, 0, stream>>>(csum, trans, dur, bp, argd, fin);
  backtrack_kernel<<<1, 32, 0, stream>>>(bp, argd, fin, (float*)d_out);
}
